// SuperGlueAttentionalPropagation_81544249082308
// MI455X (gfx1250) — compile-verified, matrix operand order checked by reading, not run on hardware
//
#include <hip/hip_runtime.h>
#include <hip/hip_bf16.h>

typedef __attribute__((ext_vector_type(16))) _Float16 v16h;
typedef __attribute__((ext_vector_type(8)))  _Float16 v8h;
typedef __attribute__((ext_vector_type(8)))  float    v8f;
typedef __attribute__((ext_vector_type(4)))  int      v4i;

constexpr int kB  = 8;
constexpr int kS  = 2048;
constexpr int kD  = 256;
constexpr int kH  = 4;
constexpr int kHD = 64;
constexpr int kMR = kB * kS;      // 16384 rows
constexpr float kEPS = 1e-5f;

static __device__ __forceinline__ v16h mk16(v8h lo, v8h hi) {
    v16h r;
#pragma unroll
    for (int i = 0; i < 8; ++i) { r[i] = lo[i]; r[i + 8] = hi[i]; }
    return r;
}

// ---------------------------------------------------------------------------
// Async global->LDS staging (CDNA5 ASYNCcnt path) with compile-safe fallback
// ---------------------------------------------------------------------------
#if defined(__gfx1250__) && __has_builtin(__builtin_amdgcn_global_load_async_to_lds_b128)
#define ASYNC_LDS 1
#else
#define ASYNC_LDS 0
#endif

static __device__ __forceinline__ void copy_b128_to_lds(const _Float16* gptr,
                                                        _Float16* lptr) {
#if ASYNC_LDS
    __builtin_amdgcn_global_load_async_to_lds_b128(
        (__attribute__((address_space(1))) v4i*)gptr,
        (__attribute__((address_space(3))) v4i*)lptr, 0, 0);
#else
    *(v8h*)lptr = *(const v8h*)gptr;
#endif
}

template <int N>
static __device__ __forceinline__ void wait_async() {
#if defined(__gfx1250__) && __has_builtin(__builtin_amdgcn_s_wait_asynccnt)
    __builtin_amdgcn_s_wait_asynccnt(N);
#elif defined(__gfx1250__)
    asm volatile("s_wait_asynccnt %0" ::"n"(N));
#endif
}

// ---------------------------------------------------------------------------
// f32 -> f16 conversion kernels
// ---------------------------------------------------------------------------
__global__ void conv_desc_to_h16(const float* __restrict__ src,
                                 _Float16* __restrict__ h16, int total) {
    int i = blockIdx.x * blockDim.x + threadIdx.x;
    if (i < total) {
        int row = i / kD, col = i % kD;
        h16[(size_t)row * (2 * kD) + col] = (_Float16)src[i];
    }
}

__global__ void conv_f16(const float* __restrict__ src,
                         _Float16* __restrict__ dst, int total) {
    int i = blockIdx.x * blockDim.x + threadIdx.x;
    if (i < total) dst[i] = (_Float16)src[i];
}

// W[inD, outD] f32 -> Wt[outD, inD] f16 (so GEMM B-operand reads are contiguous)
__global__ void transpose_w(const float* __restrict__ W,
                            _Float16* __restrict__ Wt, int inD, int outD) {
    int i = blockIdx.x * blockDim.x + threadIdx.x;
    if (i < inD * outD) {
        int r = i / outD, c = i % outD;
        Wt[(size_t)c * inD + r] = (_Float16)W[i];
    }
}

// ---------------------------------------------------------------------------
// Generic WMMA GEMM: C[M,N] = A[M,K](f16,lda) @ Wt[N,K](f16) + bias, epilogue by MODE
//   MODE 0: f16 -> [B,H,S,HD]      (Q, K projections, head-interleaved)
//   MODE 1: f16 -> [B,H,HD,S]      (V projection, pre-transposed for PV WMMA)
//   MODE 2: f16 -> h16[:, D:2D]    (attention out-proj into concat buffer)
//   MODE 3: BN+ReLU, f16 -> g16    (MLP layer 0)
//   MODE 4: f32 -> out             (MLP layer 1)
// ---------------------------------------------------------------------------
template <int MODE>
__global__ __launch_bounds__(128) void gemm_wmma(
    const _Float16* __restrict__ A, int lda,
    const _Float16* __restrict__ Wt,
    const float* __restrict__ bias,
    int M, int N, int K,
    _Float16* __restrict__ out16, float* __restrict__ out32,
    const float* __restrict__ bn_g, const float* __restrict__ bn_b,
    const float* __restrict__ bn_m, const float* __restrict__ bn_v) {
    const int lane   = threadIdx.x & 31;
    const int wv     = threadIdx.x >> 5;
    const int colL   = lane & 15;          // M (A) or N (B/C) position
    const int kc     = (lane >> 4) * 8;    // K-chunk select per ISA layout
    const int rowOff = (lane & 16) ? 8 : 0;

    const int tm = blockIdx.x * 64 + wv * 16;
    const int tn = blockIdx.y * 64;

    v8f acc[4] = {};
    const _Float16* arow = A + (size_t)(tm + colL) * lda;

    for (int kk = 0; kk < K; kk += 32) {
        v16h a = mk16(*(const v8h*)(arow + kk + kc),
                      *(const v8h*)(arow + kk + 16 + kc));
#pragma unroll
        for (int c = 0; c < 4; ++c) {
            const _Float16* wrow = Wt + (size_t)(tn + c * 16 + colL) * K + kk;
            v16h bm = mk16(*(const v8h*)(wrow + kc),
                           *(const v8h*)(wrow + 16 + kc));
            acc[c] = __builtin_amdgcn_wmma_f32_16x16x32_f16(
                false, a, false, bm, (short)0, acc[c], false, false);
        }
    }

#pragma unroll
    for (int c = 0; c < 4; ++c) {
        const int n = tn + c * 16 + colL;
        float bns = 1.f, bno = 0.f;
        if (MODE == 3) {
            float iv = rsqrtf(bn_v[n] + kEPS) * bn_g[n];
            bns = iv;
            bno = bn_b[n] - bn_m[n] * iv;
        }
        const float bv = bias[n];
#pragma unroll
        for (int r = 0; r < 8; ++r) {
            const int m = tm + r + rowOff;
            float v = acc[c][r] + bv;
            if (MODE == 0) {
                int bb = m / kS, s = m % kS, hh = n / kHD, d = n % kHD;
                out16[(((size_t)(bb * kH + hh) * kS) + s) * kHD + d] = (_Float16)v;
            } else if (MODE == 1) {
                int bb = m / kS, s = m % kS, hh = n / kHD, d = n % kHD;
                out16[(((size_t)(bb * kH + hh) * kHD) + d) * kS + s] = (_Float16)v;
            } else if (MODE == 2) {
                out16[(size_t)m * (2 * kD) + kD + n] = (_Float16)v;
            } else if (MODE == 3) {
                v = fmaxf(v * bns + bno, 0.f);
                out16[(size_t)m * (2 * kD) + n] = (_Float16)v;
            } else {
                out32[(size_t)m * kD + n] = v;
            }
        }
    }
}

// ---------------------------------------------------------------------------
// Flash attention with cooperative, double-buffered async K/V staging in LDS.
// Block = 4 waves, all on the SAME (b,h) (128 q-tiles per head, 4 per block),
// so the 32x64 K tile + 64x32 Vt tile are loaded once per block per stage via
// GLOBAL_LOAD_ASYNC_TO_LDS_B128 and consumed by all 4 waves from LDS.
// ---------------------------------------------------------------------------
__global__ __launch_bounds__(128) void flash_attn(
    const _Float16* __restrict__ q16, const _Float16* __restrict__ k16,
    const _Float16* __restrict__ vt16, const int* __restrict__ mask,
    _Float16* __restrict__ ctx16) {
    __shared__ _Float16 kbuf[2 * 32 * 64];   // [stage][krow 0..31][d 0..63]
    __shared__ _Float16 vbuf[2 * 64 * 32];   // [stage][d 0..63][krow 0..31]
    __shared__ _Float16 plds[4 * 16 * 32];   // per-wave P transpose scratch

    const int lane = threadIdx.x & 31;
    const int wv   = threadIdx.x >> 5;
    _Float16* P    = &plds[wv * 512];

    const int tile = blockIdx.x * 4 + wv;
    const int qt   = tile & (kS / 16 - 1);
    const int bh   = tile >> 7;                 // identical for all 4 waves
    const int bb   = bh / kH, hh = bh % kH;

    const int colL   = lane & 15;
    const int kc     = (lane >> 4) * 8;
    const int rowOff = (lane & 16) ? 8 : 0;

    const size_t headQ = (size_t)(bb * kH + hh) * kS;
    const _Float16* kg_base = k16 + headQ * kHD;
    const _Float16* vg_base = vt16 + (size_t)(bb * kH + hh) * kHD * kS;

    // cooperative stage issue: 128 threads x 2 b128 chunks each for K and V
    auto issue_stage = [&](int j, int buf) {
        _Float16* kb = kbuf + buf * (32 * 64);
        _Float16* vb = vbuf + buf * (64 * 32);
        const _Float16* kg = kg_base + (size_t)(j * 32) * kHD;
        const _Float16* vg = vg_base + j * 32;
#pragma unroll
        for (int i = 0; i < 2; ++i) {
            const int cid  = threadIdx.x + i * 128;   // 0..255
            const int krow = cid >> 3, kcol = (cid & 7) * 8;
            copy_b128_to_lds(kg + krow * kHD + kcol, kb + krow * 64 + kcol);
            const int vrow = cid >> 2, vcol = (cid & 3) * 8;
            copy_b128_to_lds(vg + (size_t)vrow * kS + vcol, vb + vrow * 32 + vcol);
        }
    };

    // Q tile as two 16x32 A operands (d in [0,32) and [32,64))
    const _Float16* qrow = q16 + (headQ + qt * 16 + colL) * kHD;
    const v16h aq0 = mk16(*(const v8h*)(qrow + kc), *(const v8h*)(qrow + 16 + kc));
    const v16h aq1 = mk16(*(const v8h*)(qrow + 32 + kc), *(const v8h*)(qrow + 48 + kc));

    float rmax[8], rsum[8];
    v8f acc[4] = {};
#pragma unroll
    for (int r = 0; r < 8; ++r) { rmax[r] = -1e30f; rsum[r] = 0.f; }

    issue_stage(0, 0);

    for (int j = 0; j < kS / 32; ++j) {
        const int cur = j & 1;
        if (j + 1 < kS / 32) {
            issue_stage(j + 1, cur ^ 1);  // overlap next transfers with compute
            wait_async<4>();              // in-order: stage j (4 ops/wave) done
        } else {
            wait_async<0>();
        }
        __syncthreads();                  // all waves' stage-j data visible

        const _Float16* kb = kbuf + cur * (32 * 64);
        const _Float16* vb = vbuf + cur * (64 * 32);

        float s0[8], s1[8];
#pragma unroll
        for (int t = 0; t < 2; ++t) {
            const _Float16* krow = kb + (t * 16 + colL) * 64;
            v16h b0 = mk16(*(const v8h*)(krow + kc), *(const v8h*)(krow + 16 + kc));
            v16h b1 = mk16(*(const v8h*)(krow + 32 + kc), *(const v8h*)(krow + 48 + kc));
            v8f c = {};
            c = __builtin_amdgcn_wmma_f32_16x16x32_f16(false, aq0, false, b0, (short)0, c, false, false);
            c = __builtin_amdgcn_wmma_f32_16x16x32_f16(false, aq1, false, b1, (short)0, c, false, false);
            const int mk = mask[bb * kS + j * 32 + t * 16 + colL];
            float* sd = t ? s1 : s0;
#pragma unroll
            for (int r = 0; r < 8; ++r) sd[r] = mk ? c[r] * 0.125f : -1e9f;
        }

        // online softmax over the 32 new columns (row r spread over 16 lanes)
#pragma unroll
        for (int r = 0; r < 8; ++r) {
            float m2 = fmaxf(s0[r], s1[r]);
            m2 = fmaxf(m2, __shfl_xor(m2, 1, 32));
            m2 = fmaxf(m2, __shfl_xor(m2, 2, 32));
            m2 = fmaxf(m2, __shfl_xor(m2, 4, 32));
            m2 = fmaxf(m2, __shfl_xor(m2, 8, 32));
            const float nm    = fmaxf(rmax[r], m2);
            const float alpha = __expf(rmax[r] - nm);
            rmax[r] = nm;
            const float p0 = __expf(s0[r] - nm);
            const float p1 = __expf(s1[r] - nm);
            float ps = p0 + p1;
            ps += __shfl_xor(ps, 1, 32);
            ps += __shfl_xor(ps, 2, 32);
            ps += __shfl_xor(ps, 4, 32);
            ps += __shfl_xor(ps, 8, 32);
            rsum[r] = rsum[r] * alpha + ps;
#pragma unroll
            for (int c = 0; c < 4; ++c) acc[c][r] *= alpha;
            s0[r] = p0;
            s1[r] = p1;
        }

        // P: C-layout -> row-major [16][32] in LDS (transpose for A-operand)
#pragma unroll
        for (int r = 0; r < 8; ++r) {
            const int m = r + rowOff;
            P[m * 32 + colL]      = (_Float16)s0[r];
            P[m * 32 + 16 + colL] = (_Float16)s1[r];
        }
        __syncthreads();
        const _Float16* prow = P + colL * 32;
        const v16h pa = mk16(*(const v8h*)(prow + kc), *(const v8h*)(prow + 16 + kc));

        // PV: ctx[16x64] += P[16x32] @ V[32x64] from LDS Vt
#pragma unroll
        for (int c = 0; c < 4; ++c) {
            const _Float16* vrow = vb + (c * 16 + colL) * 32;
            v16h vbm = mk16(*(const v8h*)(vrow + kc), *(const v8h*)(vrow + 16 + kc));
            acc[c] = __builtin_amdgcn_wmma_f32_16x16x32_f16(
                false, pa, false, vbm, (short)0, acc[c], false, false);
        }
        __syncthreads();  // all reads of kb/vb/P done; next stage may overwrite
    }

    // normalize and store ctx back in [B,S,D] f16 layout (head-deinterleaved)
#pragma unroll
    for (int r = 0; r < 8; ++r) rsum[r] = 1.0f / rsum[r];
#pragma unroll
    for (int c = 0; c < 4; ++c)
#pragma unroll
        for (int r = 0; r < 8; ++r) {
            const int s   = qt * 16 + r + rowOff;
            const int col = hh * kHD + c * 16 + colL;
            ctx16[((size_t)bb * kS + s) * kD + col] = (_Float16)(acc[c][r] * rsum[r]);
        }
}

// ---------------------------------------------------------------------------
extern "C" void kernel_launch(void* const* d_in, const int* in_sizes, int n_in,
                              void* d_out, int out_size, void* d_ws, size_t ws_size,
                              hipStream_t stream) {
    const float* desc = (const float*)d_in[0];
    const float* srcp = (const float*)d_in[1];
    const int*   mask = (const int*)d_in[2];
    const float* q_w = (const float*)d_in[3];  const float* q_b = (const float*)d_in[4];
    const float* k_w = (const float*)d_in[5];  const float* k_b = (const float*)d_in[6];
    const float* v_w = (const float*)d_in[7];  const float* v_b = (const float*)d_in[8];
    const float* o_w = (const float*)d_in[9];  const float* o_b = (const float*)d_in[10];
    const float* m0_w = (const float*)d_in[11]; const float* m0_b = (const float*)d_in[12];
    const float* bn_g = (const float*)d_in[13]; const float* bn_b = (const float*)d_in[14];
    const float* bn_m = (const float*)d_in[15]; const float* bn_v = (const float*)d_in[16];
    const float* m1_w = (const float*)d_in[17]; const float* m1_b = (const float*)d_in[18];

    char* ws = (char*)d_ws;
    size_t off = 0;
    auto alloc = [&](size_t bytes) -> void* {
        void* p = ws + off;
        off += (bytes + 255) & ~(size_t)255;
        return p;
    };

    _Float16* src16 = (_Float16*)alloc((size_t)kMR * kD * 2);
    _Float16* q16   = (_Float16*)alloc((size_t)kMR * kD * 2);
    _Float16* kk16  = (_Float16*)alloc((size_t)kMR * kD * 2);
    _Float16* vt16  = (_Float16*)alloc((size_t)kMR * kD * 2);
    _Float16* ctx16 = (_Float16*)alloc((size_t)kMR * kD * 2);
    _Float16* h16   = (_Float16*)alloc((size_t)kMR * 2 * kD * 2);
    _Float16* g16   = (_Float16*)alloc((size_t)kMR * 2 * kD * 2);
    _Float16* wq  = (_Float16*)alloc((size_t)kD * kD * 2);
    _Float16* wk  = (_Float16*)alloc((size_t)kD * kD * 2);
    _Float16* wvv = (_Float16*)alloc((size_t)kD * kD * 2);
    _Float16* wo  = (_Float16*)alloc((size_t)kD * kD * 2);
    _Float16* wm0 = (_Float16*)alloc((size_t)2 * kD * 2 * kD * 2);
    _Float16* wm1 = (_Float16*)alloc((size_t)2 * kD * kD * 2);

    const int tE = kMR * kD;
    conv_desc_to_h16<<<(tE + 255) / 256, 256, 0, stream>>>(desc, h16, tE);
    conv_f16<<<(tE + 255) / 256, 256, 0, stream>>>(srcp, src16, tE);

    transpose_w<<<(kD * kD + 255) / 256, 256, 0, stream>>>(q_w, wq, kD, kD);
    transpose_w<<<(kD * kD + 255) / 256, 256, 0, stream>>>(k_w, wk, kD, kD);
    transpose_w<<<(kD * kD + 255) / 256, 256, 0, stream>>>(v_w, wvv, kD, kD);
    transpose_w<<<(kD * kD + 255) / 256, 256, 0, stream>>>(o_w, wo, kD, kD);
    transpose_w<<<(4 * kD * kD + 255) / 256, 256, 0, stream>>>(m0_w, wm0, 2 * kD, 2 * kD);
    transpose_w<<<(2 * kD * kD + 255) / 256, 256, 0, stream>>>(m1_w, wm1, 2 * kD, kD);

    dim3 blk(128);
    gemm_wmma<0><<<dim3(kMR / 64, kD / 64), blk, 0, stream>>>(
        h16, 2 * kD, wq, q_b, kMR, kD, kD, q16, nullptr, nullptr, nullptr, nullptr, nullptr);
    gemm_wmma<0><<<dim3(kMR / 64, kD / 64), blk, 0, stream>>>(
        src16, kD, wk, k_b, kMR, kD, kD, kk16, nullptr, nullptr, nullptr, nullptr, nullptr);
    gemm_wmma<1><<<dim3(kMR / 64, kD / 64), blk, 0, stream>>>(
        src16, kD, wvv, v_b, kMR, kD, kD, vt16, nullptr, nullptr, nullptr, nullptr, nullptr);

    flash_attn<<<kB * kH * (kS / 16) / 4, 128, 0, stream>>>(q16, kk16, vt16, mask, ctx16);

    gemm_wmma<2><<<dim3(kMR / 64, kD / 64), blk, 0, stream>>>(
        ctx16, kD, wo, o_b, kMR, kD, kD, h16, nullptr, nullptr, nullptr, nullptr, nullptr);
    gemm_wmma<3><<<dim3(kMR / 64, (2 * kD) / 64), blk, 0, stream>>>(
        h16, 2 * kD, wm0, m0_b, kMR, 2 * kD, 2 * kD, g16, nullptr, bn_g, bn_b, bn_m, bn_v);
    gemm_wmma<4><<<dim3(kMR / 64, kD / 64), blk, 0, stream>>>(
        g16, 2 * kD, wm1, m1_b, kMR, kD, 2 * kD, nullptr, (float*)d_out,
        nullptr, nullptr, nullptr, nullptr);
}
